// GAE_77979426226957
// MI455X (gfx1250) — compile-verified, matrix operand order checked by reading, not run on hardware
//
#include <hip/hip_runtime.h>
#include <hip/hip_bf16.h>

// ---------------------------------------------------------------------------
// Types for CDNA5 WMMA
// ---------------------------------------------------------------------------
typedef __attribute__((ext_vector_type(16))) __bf16 v16bf;
typedef __attribute__((ext_vector_type(8)))  float  v8f;

#define TM 128
#define TN 128
#define TK 32
#define LDK (TK + 24)   // bf16 stride 56 -> 112B: 16B-aligned, bank-staggered

// ---------------------------------------------------------------------------
// fp32 -> bf16 conversion
// ---------------------------------------------------------------------------
__global__ __launch_bounds__(256)
void f32_to_bf16(const float* __restrict__ in, __bf16* __restrict__ out, int n) {
  int i = blockIdx.x * 256 + threadIdx.x;
  if (i < n) out[i] = (__bf16)in[i];
}

// ---------------------------------------------------------------------------
// WMMA fragment load from LDS following ISA 16-bit A-matrix 16x32 layout:
// lane = hf*16 + row16 ; VGPR v (pair) holds K = (v>=4?16:0) + hf*8 + (v&3)*2
// (B fragment uses the mirrored layout with N = lane&15.)
// ---------------------------------------------------------------------------
__device__ __forceinline__ v16bf lds_frag(const __bf16* base, int row, int hf) {
  v16bf f;
#pragma unroll
  for (int v = 0; v < 8; ++v) {
    int k = ((v & 4) << 2) + hf * 8 + (v & 3) * 2;
    const __bf16* p = base + row * LDK + k;
    f[2 * v]     = p[0];
    f[2 * v + 1] = p[1];
  }
  return f;
}

// ---------------------------------------------------------------------------
// Async global -> LDS staging of one 128x32 bf16 tile chunk.
// Each thread issues two GLOBAL_LOAD_ASYNC_TO_LDS_B128 (16B each):
// 512 x 16B = 8KB per tile, tracked with ASYNCcnt.
// lds_off: byte offset of the destination buffer within LDS.
// ---------------------------------------------------------------------------
__device__ __forceinline__ void async_stage(const __bf16* __restrict__ g, int ldg,
                                            int rowbase, int k0,
                                            unsigned lds_off, int t)
{
#pragma unroll
  for (int rep = 0; rep < 2; ++rep) {
    int i = t + rep * 256;           // 0..511
    int row = i >> 2;                // 4 x 16B chunks per 32-elem row
    int cc  = (i & 3) * 8;
    const __bf16* gp = g + (size_t)(rowbase + row) * ldg + k0 + cc;
    unsigned lp = lds_off + (unsigned)((row * LDK + cc) * 2);
    asm volatile("global_load_async_to_lds_b128 %0, %1, off"
                 :: "v"(lp), "v"(gp) : "memory");
  }
}

// ---------------------------------------------------------------------------
// C[M,Nn] = A[M,K] * B[Nn,K]^T  (bf16 in, f32 accumulate)
// epilogue: 0 = none, 1 = sigmoid + non-temporal store
// Requires M,Nn multiples of 128 and K multiple of 32 (true for all calls).
// Block: 256 threads = 8 wave32; block tile 128x128; per-wave 32x64 (2x4 WMMA).
// Double-buffered LDS, async global->LDS prefetch one K-step ahead.
// ---------------------------------------------------------------------------
__global__ __launch_bounds__(256)
void gemm_nt_bf16(const __bf16* __restrict__ A, const __bf16* __restrict__ B,
                  float* __restrict__ C, int M, int Nn, int K, int epilogue)
{
  (void)M;
  __shared__ __bf16 lsa[2][TM * LDK];
  __shared__ __bf16 lsb[2][TN * LDK];

  const int rb = blockIdx.y * TM;
  const int cb = blockIdx.x * TN;
  const int t  = threadIdx.x;
  const int w  = t >> 5;       // wave 0..7
  const int l  = t & 31;
  const int hf = l >> 4;       // lane half
  const int ln = l & 15;
  const int mw = (w & 3) * 32; // wave M offset inside tile
  const int nw = (w >> 2) * 64;// wave N offset inside tile

  const unsigned abase = (unsigned)(uintptr_t)&lsa[0][0];
  const unsigned bbase = (unsigned)(uintptr_t)&lsb[0][0];
  const unsigned tile_bytes = (unsigned)(TM * LDK * 2);

  v8f acc[2][4];
#pragma unroll
  for (int i = 0; i < 2; ++i)
#pragma unroll
    for (int j = 0; j < 4; ++j) acc[i][j] = (v8f)(0.0f);

  // prologue: chunk 0 -> buffer 0 (4 async b128 loads per thread)
  async_stage(A, K, rb, 0, abase, t);
  async_stage(B, K, cb, 0, bbase, t);

  const int S = K / TK;
  for (int s = 0; s < S; ++s) {
    const int cur = s & 1;
    if (s + 1 < S) {
      const unsigned nxt = (unsigned)((s + 1) & 1) * tile_bytes;
      async_stage(A, K, rb, (s + 1) * TK, abase + nxt, t);
      async_stage(B, K, cb, (s + 1) * TK, bbase + nxt, t);
      // next chunk (4 per thread) may stay in flight; current must be done
      asm volatile("s_wait_asynccnt 0x4" ::: "memory");
    } else {
      asm volatile("s_wait_asynccnt 0x0" ::: "memory");
    }
    __syncthreads();

    const __bf16* pa = &lsa[cur][0];
    const __bf16* pb = &lsb[cur][0];
    v16bf af[2], bfr[4];
#pragma unroll
    for (int i = 0; i < 2; ++i) af[i]  = lds_frag(pa, mw + i * 16 + ln, hf);
#pragma unroll
    for (int j = 0; j < 4; ++j) bfr[j] = lds_frag(pb, nw + j * 16 + ln, hf);

#pragma unroll
    for (int i = 0; i < 2; ++i)
#pragma unroll
      for (int j = 0; j < 4; ++j)
        acc[i][j] = __builtin_amdgcn_wmma_f32_16x16x32_bf16(
            false, af[i], false, bfr[j], (short)0, acc[i][j], false, false);

    if (s + 1 < S) __syncthreads();  // protect buffer overwritten next issue
  }

  // C/D layout: VGPR r, lane -> (M = hf*8 + r, N = ln)
#pragma unroll
  for (int i = 0; i < 2; ++i)
#pragma unroll
    for (int j = 0; j < 4; ++j)
#pragma unroll
      for (int r = 0; r < 8; ++r) {
        int row = rb + mw + i * 16 + hf * 8 + r;
        int col = cb + nw + j * 16 + ln;
        float v = acc[i][j][r];
        if (epilogue) {
          v = 1.0f / (1.0f + __expf(-v));
          __builtin_nontemporal_store(v, &C[(size_t)row * Nn + col]);
        } else {
          C[(size_t)row * Nn + col] = v;
        }
      }
}

// ---------------------------------------------------------------------------
// Edge kernels: segment softmax + aggregation (self-loop idx >= E -> node)
// ---------------------------------------------------------------------------
__global__ __launch_bounds__(256)
void init_softmax(unsigned* __restrict__ m, float* __restrict__ s, int n) {
  int i = blockIdx.x * 256 + threadIdx.x;
  if (i < n) { m[i] = 0xFF800000u; s[i] = 0.0f; }   // -inf bits, 0
}

__global__ __launch_bounds__(256)
void zero_f32(float* __restrict__ p, int n) {
  int i = blockIdx.x * 256 + threadIdx.x;
  if (i < n) p[i] = 0.0f;
}

__device__ __forceinline__ void atomicMaxF(unsigned* addr, float v) {
  int vi = __float_as_int(v);
  if (vi >= 0) atomicMax((int*)addr, vi);
  else         atomicMin(addr, (unsigned)vi);
}

__global__ __launch_bounds__(256)
void edge_logits(const float* __restrict__ xl, const float* __restrict__ xr,
                 const float* __restrict__ att,
                 const int* __restrict__ src, const int* __restrict__ dst,
                 int E, int N, int OC,
                 float* __restrict__ e_out, unsigned* __restrict__ m)
{
  int idx = blockIdx.x * 256 + threadIdx.x;
  int Etot = E + N;
  if (idx >= Etot) return;
  int s = (idx < E) ? src[idx] : (idx - E);
  int d = (idx < E) ? dst[idx] : (idx - E);
  const float4* pl = (const float4*)(xl + (size_t)s * OC);
  const float4* pr = (const float4*)(xr + (size_t)d * OC);
  const float4* pa = (const float4*)att;
  float acc = 0.0f;
  int nc = OC >> 2;
  for (int c = 0; c < nc; ++c) {
    float4 a = pl[c], b = pr[c], wv = pa[c];
    float v;
    v = a.x + b.x; acc += wv.x * (v > 0.0f ? v : 0.2f * v);
    v = a.y + b.y; acc += wv.y * (v > 0.0f ? v : 0.2f * v);
    v = a.z + b.z; acc += wv.z * (v > 0.0f ? v : 0.2f * v);
    v = a.w + b.w; acc += wv.w * (v > 0.0f ? v : 0.2f * v);
  }
  e_out[idx] = acc;
  atomicMaxF(m + d, acc);
}

__global__ __launch_bounds__(256)
void edge_p(const float* __restrict__ e, const unsigned* __restrict__ m,
            const int* __restrict__ dst, int E, int N,
            float* __restrict__ p, float* __restrict__ s)
{
  int idx = blockIdx.x * 256 + threadIdx.x;
  int Etot = E + N;
  if (idx >= Etot) return;
  int d = (idx < E) ? dst[idx] : (idx - E);
  float pv = __expf(e[idx] - __uint_as_float(m[d]));
  p[idx] = pv;
  atomicAdd(s + d, pv);
}

// one 8-edge row-group per block; 32 lanes cover channels with float4
__global__ __launch_bounds__(256)
void edge_aggregate(const float* __restrict__ xl, const float* __restrict__ p,
                    const float* __restrict__ s,
                    const int* __restrict__ src, const int* __restrict__ dst,
                    int E, int N, int OC, float* __restrict__ out)
{
  int idx = blockIdx.x * blockDim.y + threadIdx.y;
  int Etot = E + N;
  if (idx >= Etot) return;
  int sn = (idx < E) ? src[idx] : (idx - E);
  int dn = (idx < E) ? dst[idx] : (idx - E);
  float alpha = p[idx] / s[dn];
  for (int c = threadIdx.x * 4; c < OC; c += 128) {
    float4 v = *(const float4*)(xl + (size_t)sn * OC + c);
    float* o = out + (size_t)dn * OC + c;
    atomicAdd(o + 0, alpha * v.x);
    atomicAdd(o + 1, alpha * v.y);
    atomicAdd(o + 2, alpha * v.z);
    atomicAdd(o + 3, alpha * v.w);
  }
}

__global__ __launch_bounds__(256)
void bias_relu(float* __restrict__ out, const float* __restrict__ b,
               int n, int ocmask)
{
  int i = blockIdx.x * 256 + threadIdx.x;
  if (i < n) {
    float v = out[i] + b[i & ocmask];
    out[i] = v > 0.0f ? v : 0.0f;
  }
}

// ---------------------------------------------------------------------------
// Host orchestration
// ---------------------------------------------------------------------------
static inline int cdiv(int a, int b) { return (a + b - 1) / b; }

namespace {
struct Scratch {
  __bf16 *xb, *wlb, *wrb;
  float *xl, *xr, *e, *p, *s;
  unsigned* m;
};
}

static void run_layer(const float* xin, int IC, int OC,
                      const float* wl, const float* wr,
                      const float* att, const float* bias,
                      const int* src, const int* dst, int E, int N,
                      float* outp, const Scratch& S, hipStream_t stream)
{
  int Etot = E + N;
  f32_to_bf16<<<cdiv(N * IC, 256), 256, 0, stream>>>(xin, S.xb, N * IC);
  f32_to_bf16<<<cdiv(OC * IC, 256), 256, 0, stream>>>(wl, S.wlb, OC * IC);
  f32_to_bf16<<<cdiv(OC * IC, 256), 256, 0, stream>>>(wr, S.wrb, OC * IC);

  dim3 g(OC / TN, N / TM);
  gemm_nt_bf16<<<g, 256, 0, stream>>>(S.xb, S.wlb, S.xl, N, OC, IC, 0);
  gemm_nt_bf16<<<g, 256, 0, stream>>>(S.xb, S.wrb, S.xr, N, OC, IC, 0);

  init_softmax<<<cdiv(N, 256), 256, 0, stream>>>(S.m, S.s, N);
  zero_f32<<<cdiv(N * OC, 256), 256, 0, stream>>>(outp, N * OC);

  edge_logits<<<cdiv(Etot, 256), 256, 0, stream>>>(S.xl, S.xr, att, src, dst,
                                                   E, N, OC, S.e, S.m);
  edge_p<<<cdiv(Etot, 256), 256, 0, stream>>>(S.e, S.m, dst, E, N, S.p, S.s);

  dim3 b(32, 8);
  edge_aggregate<<<cdiv(Etot, 8), b, 0, stream>>>(S.xl, S.p, S.s, src, dst,
                                                  E, N, OC, outp);
  bias_relu<<<cdiv(N * OC, 256), 256, 0, stream>>>(outp, bias, N * OC, OC - 1);
}

extern "C" void kernel_launch(void* const* d_in, const int* in_sizes, int n_in,
                              void* d_out, int out_size, void* d_ws, size_t ws_size,
                              hipStream_t stream)
{
  (void)n_in; (void)out_size; (void)ws_size;
  const int N = 8192, IN_CH = 256, CH = 128;

  const float* x  = (const float*)d_in[0];
  const int*   ei = (const int*)d_in[1];
  const int    E  = in_sizes[1] / 2;
  const int* src = ei;
  const int* dst = ei + E;

  const float* P[5][4];  // wl, wr, att, bias per layer
  for (int i = 0; i < 5; ++i)
    for (int j = 0; j < 4; ++j)
      P[i][j] = (const float*)d_in[2 + i * 4 + j];

  float* out_recon = (float*)d_out;
  float* out_xrec  = out_recon + (size_t)N * N;
  float* out_z     = out_xrec + (size_t)N * IN_CH;

  // workspace carve (256B aligned)
  char* wsp = (char*)d_ws;
  auto take = [&](size_t bytes) -> void* {
    void* p = (void*)wsp;
    wsp += (bytes + 255) & ~(size_t)255;
    return p;
  };
  Scratch S;
  S.xb  = (__bf16*)take((size_t)N * 256 * 2);
  S.wlb = (__bf16*)take((size_t)256 * 256 * 2);
  S.wrb = (__bf16*)take((size_t)256 * 256 * 2);
  S.xl  = (float*)take((size_t)N * 256 * 4);
  S.xr  = (float*)take((size_t)N * 256 * 4);
  int Etot = E + N;
  S.e   = (float*)take((size_t)Etot * 4);
  S.p   = (float*)take((size_t)Etot * 4);
  S.m   = (unsigned*)take((size_t)N * 4);
  S.s   = (float*)take((size_t)N * 4);
  float* h1  = (float*)take((size_t)N * CH * 4);
  float* h3  = (float*)take((size_t)N * CH * 4);
  float* h4  = (float*)take((size_t)N * CH * 4);
  __bf16* reb = (__bf16*)take((size_t)N * CH * 2);

  // layer 1: [N,256] -> [N,128]
  run_layer(x, IN_CH, CH, P[0][0], P[0][1], P[0][2], P[0][3],
            src, dst, E, N, h1, S, stream);
  // layer 2: [N,128] -> [N,256] = z  (written straight into d_out)
  run_layer(h1, CH, 2 * CH, P[1][0], P[1][1], P[1][2], P[1][3],
            src, dst, E, N, out_z, S, stream);
  // layer 3: z -> re [N,128]
  run_layer(out_z, 2 * CH, CH, P[2][0], P[2][1], P[2][2], P[2][3],
            src, dst, E, N, h3, S, stream);
  // layer 4: z -> xd [N,128]
  run_layer(out_z, 2 * CH, CH, P[3][0], P[3][1], P[3][2], P[3][3],
            src, dst, E, N, h4, S, stream);
  // layer 5: xd -> x_rec [N,256] (written straight into d_out)
  run_layer(h4, CH, IN_CH, P[4][0], P[4][1], P[4][2], P[4][3],
            src, dst, E, N, out_xrec, S, stream);

  // recon_edge = sigmoid(re @ re^T): 8192x8192x128 bf16 WMMA GEMM,
  // fused sigmoid + non-temporal stores (268MB write-once output)
  f32_to_bf16<<<cdiv(N * CH, 256), 256, 0, stream>>>(h3, reb, N * CH);
  dim3 gr(N / TN, N / TM);
  gemm_nt_bf16<<<gr, 256, 0, stream>>>(reb, reb, out_recon, N, N, CH, 1);
}
